// PAM_13898514170484
// MI455X (gfx1250) — matrix operand order repaired and checked by reading, not compiled, not run on hardware
//
#include <hip/hip_runtime.h>

typedef __attribute__((ext_vector_type(16))) _Float16 v16h;
typedef __attribute__((ext_vector_type(8)))  _Float16 v8h;
typedef __attribute__((ext_vector_type(8)))  float    v8f;
typedef unsigned int u32x4 __attribute__((ext_vector_type(4)));
typedef int          i32x4 __attribute__((ext_vector_type(4)));
typedef int          i32x8 __attribute__((ext_vector_type(8)));

#define CC 64
#define HW 4096
#define NT 32    // n-chunk width per iteration
#define MB 128   // m-columns per block (4 waves x 2 subtiles x 16)

#if __has_builtin(__builtin_amdgcn_tensor_load_to_lds) && __has_builtin(__builtin_amdgcn_s_wait_tensorcnt)
#define HAVE_TDM 1
#endif

#ifdef HAVE_TDM
// TDM 2D tile load: tile_dim0=NT f32 per row, tile_dim1=CC rows, row stride HW, -> contiguous LDS
__device__ __forceinline__ void tdm_load_tile(const float* gsrc, unsigned lds_off) {
    unsigned long long ga = (unsigned long long)(size_t)gsrc;
    u32x4 g0;
    g0[0] = 1u;                                         // count=1 (valid), flags=0
    g0[1] = lds_off;                                    // D#.lds_addr (bytes)
    g0[2] = (unsigned)(ga & 0xFFFFFFFFu);               // global_addr[31:0]
    g0[3] = (unsigned)((ga >> 32) & 0x01FFFFFFu)        // global_addr[56:32]
          | (2u << 30);                                 // type=2 ("image")
    i32x8 g1;
    g1[0] = (int)(2u << 16);                            // data_size=2 -> 4 bytes
    g1[1] = (int)((unsigned)HW << 16);                  // tensor_dim0[15:0]=4096
    g1[2] = (int)(((unsigned)HW >> 16) | (64u << 16));  // dim0[31:16] | tensor_dim1[15:0]=64
    g1[3] = (int)((unsigned)NT << 16);                  // dim1[31:16]=0 | tile_dim0=32
    g1[4] = (int)(unsigned)CC;                          // tile_dim1=64, tile_dim2=0
    g1[5] = (int)(unsigned)HW;                          // tensor_dim0_stride[31:0]=4096
    g1[6] = 0;                                          // stride0 hi | stride1 lo
    g1[7] = 0;                                          // stride1 hi
    i32x4 z4 = {0, 0, 0, 0};
#if __clang_major__ >= 23
    i32x8 z8 = {0, 0, 0, 0, 0, 0, 0, 0};
    __builtin_amdgcn_tensor_load_to_lds(g0, g1, z4, z4, z8, 0);
#else
    __builtin_amdgcn_tensor_load_to_lds(g0, g1, z4, z4, 0);
#endif
}
#endif

// ---------------- Kernel 1: M_b = max_n sum_c f[c,n]^2 ----------------
__global__ __launch_bounds__(256) void pam_rowmax(const float* __restrict__ img,
                                                  float* __restrict__ Mmax) {
    __shared__ float red[256];
    const int bi = blockIdx.y;
    const int n  = blockIdx.x * 256 + threadIdx.x;
    const float* f = img + (size_t)bi * CC * HW;
    float s = 0.f;
#pragma unroll 8
    for (int c = 0; c < CC; ++c) {
        float v = f[c * HW + n];
        s = fmaf(v, v, s);
    }
    red[threadIdx.x] = s;
    __syncthreads();
    for (int off = 128; off > 0; off >>= 1) {
        if (threadIdx.x < off) red[threadIdx.x] = fmaxf(red[threadIdx.x], red[threadIdx.x + off]);
        __syncthreads();
    }
    if (threadIdx.x == 0)
        atomicMax((unsigned int*)&Mmax[bi], __float_as_uint(red[0]));  // vals >= 0
}

// ---------------- Kernel 2: fused E=exp(S-M); outU = f*E; Z partials ----------------
__global__ __launch_bounds__(128) void pam_main(const float* __restrict__ img,
                                                float* __restrict__ out,
                                                const float* __restrict__ Mmax,
                                                float* __restrict__ zpart) {
    __shared__ alignas(32) _Float16 B1frag[8 * 2 * 32 * 16];  // [msub][kt] GEMM1-B, 16KB
    __shared__ alignas(16) float    fnraw32[2][CC * NT];      // TDM dest, double-buffered, 16KB
    __shared__ alignas(32) _Float16 A1frag[4 * 32 * 16];      // [ns*2+kt]  GEMM1-A, 4KB
    __shared__ alignas(32) _Float16 A2frag[4 * 32 * 16];      // [ct]       GEMM2-A, 4KB
    __shared__ alignas(32) _Float16 EBfrag[8 * 32 * 16];      // [wid*2+j]  GEMM2-B, 8KB
    __shared__ float zred[128];

    const int tid  = threadIdx.x;
    const int lane = tid & 31;
    const int wid  = tid >> 5;
    const int half = lane >> 4;
    const int l16  = lane & 15;
    const int bi   = blockIdx.y;
    const int m0   = blockIdx.x * MB;

    const float* f = img + (size_t)bi * CC * HW;
    const float  Mb = Mmax[bi];

    // ---- one-time: GEMM1 B fragments (16 consecutive c -> 16 contiguous slots) ----
#pragma unroll
    for (int i = 0; i < 4; ++i) {
        int r  = i * 128 + tid;
        int m  = r & 127;
        int g  = r >> 7;
        int kt = g >> 1, hf = g & 1;
        const float* src = f + (kt * 32 + hf * 16) * HW + m0 + m;
        v16h pk;
#pragma unroll
        for (int s = 0; s < 16; ++s) pk[s] = (_Float16)src[s * HW];
        *(v16h*)&B1frag[(((m >> 4) * 2 + kt) * 32 + hf * 16 + (m & 15)) * 16] = pk;
    }

    v8f acc[8];
#pragma unroll
    for (int i = 0; i < 8; ++i) acc[i] = (v8f){};
    float zloc = 0.f;

    const int nIter = HW / NT;
#ifdef HAVE_TDM
    if (wid == 0) tdm_load_tile(f, (unsigned)(size_t)&fnraw32[0][0]);   // prefetch chunk 0
#endif

    for (int it = 0; it < nIter; ++it) {
        const int n0  = it * NT;
        const int buf = it & 1;
#ifdef HAVE_TDM
        if (wid == 0) __builtin_amdgcn_s_wait_tensorcnt(0);   // chunk `it` landed
        __syncthreads();                                      // visible to all; buf^1 readers drained
        if (wid == 0 && it + 1 < nIter)                       // overlap DMA of next chunk
            tdm_load_tile(f + (size_t)(n0 + NT), (unsigned)(size_t)&fnraw32[buf ^ 1][0]);
#else
        __syncthreads();
#pragma unroll
        for (int i = 0; i < 2; ++i) {                          // manual fallback staging
            int q = i * 128 + tid;
            int c = q >> 2, k0 = (q & 3) * 8;
            const float* src = f + c * HW + n0 + k0;
            *(float4*)&fnraw32[buf][c * NT + k0]     = *(const float4*)src;
            *(float4*)&fnraw32[buf][c * NT + k0 + 4] = *(const float4*)(src + 4);
        }
        __syncthreads();
#endif
        // ---- phase B: fnraw32[buf] (f32) -> f16 operand fragments ----
#pragma unroll
        for (int i = 0; i < 2; ++i) {
            int q = i * 128 + tid;
            // A2 (GEMM2 A, M=c K=n): 8 consecutive k -> contiguous slots
            int c = q >> 2, k0 = (q & 3) * 8;
            const float* sp = &fnraw32[buf][c * NT + k0];
            float4 a = *(const float4*)sp;
            float4 b = *(const float4*)(sp + 4);
            v8h pk;
            pk[0] = (_Float16)a.x; pk[1] = (_Float16)a.y; pk[2] = (_Float16)a.z; pk[3] = (_Float16)a.w;
            pk[4] = (_Float16)b.x; pk[5] = (_Float16)b.y; pk[6] = (_Float16)b.z; pk[7] = (_Float16)b.w;
            int hf = (k0 >> 4) & 1, sb = (k0 & 8) ? 8 : 0;
            *(v8h*)&A2frag[(((c >> 4) * 32) + hf * 16 + (c & 15)) * 16 + sb] = pk;
            // A1 (GEMM1 A, M=n K=c): 8 consecutive c -> contiguous slots (transpose gather)
            int k = q & 31, c0 = (q >> 5) * 8;
            v8h pk2;
#pragma unroll
            for (int t = 0; t < 8; ++t) pk2[t] = (_Float16)fnraw32[buf][(c0 + t) * NT + k];
            int ns = k >> 4, kt = c0 >> 5;
            int hf2 = (c0 >> 4) & 1, sb2 = (c0 & 8) ? 8 : 0;
            *(v8h*)&A1frag[(((ns * 2 + kt) * 32) + hf2 * 16 + (k & 15)) * 16 + sb2] = pk2;
        }
        __syncthreads();

        // ---- GEMM1: S = f^T f (K=64); exp; pack E straight into B-operand fragments ----
#pragma unroll
        for (int j = 0; j < 2; ++j) {
            const int msub = wid * 2 + j;
            v16h B0 = *(const v16h*)&B1frag[((msub * 2 + 0) * 32 + lane) * 16];
            v16h B1 = *(const v16h*)&B1frag[((msub * 2 + 1) * 32 + lane) * 16];
#pragma unroll
            for (int ns = 0; ns < 2; ++ns) {
                v16h A0  = *(const v16h*)&A1frag[((ns * 2 + 0) * 32 + lane) * 16];
                v16h A1v = *(const v16h*)&A1frag[((ns * 2 + 1) * 32 + lane) * 16];
                v8f s_acc = (v8f){};
                s_acc = __builtin_amdgcn_wmma_f32_16x16x32_f16(false, A0,  false, B0, (short)0, s_acc, false, false);
                s_acc = __builtin_amdgcn_wmma_f32_16x16x32_f16(false, A1v, false, B1, (short)0, s_acc, false, false);
                v8h e8;
#pragma unroll
                for (int r = 0; r < 8; ++r) {
                    float e = __expf(s_acc[r] - Mb);   // in (0,1] -> f16-safe
                    zloc += e;
                    e8[r] = (_Float16)e;
                }
                *(v8h*)&EBfrag[((wid * 2 + j) * 32 + ns * 16 + l16) * 16 + half * 8] = e8;
            }
        }
        __syncthreads();   // E cross-lane visibility

        // ---- GEMM2: acc[c,m] += f[c,nchunk] * E (K=32) ----
        v16h BE0 = *(const v16h*)&EBfrag[((wid * 2 + 0) * 32 + lane) * 16];
        v16h BE1 = *(const v16h*)&EBfrag[((wid * 2 + 1) * 32 + lane) * 16];
#pragma unroll
        for (int ct = 0; ct < 4; ++ct) {
            v16h A = *(const v16h*)&A2frag[(ct * 32 + lane) * 16];
            acc[0 * 4 + ct] = __builtin_amdgcn_wmma_f32_16x16x32_f16(false, A, false, BE0, (short)0, acc[0 * 4 + ct], false, false);
            acc[1 * 4 + ct] = __builtin_amdgcn_wmma_f32_16x16x32_f16(false, A, false, BE1, (short)0, acc[1 * 4 + ct], false, false);
        }
    }

    // ---- write unnormalized output ----
#pragma unroll
    for (int j = 0; j < 2; ++j) {
        int m = m0 + (wid * 2 + j) * 16 + l16;
#pragma unroll
        for (int ct = 0; ct < 4; ++ct) {
            v8f ac = acc[j * 4 + ct];
#pragma unroll
            for (int r = 0; r < 8; ++r) {
                int c = ct * 16 + half * 8 + r;   // D layout: VGPR r -> M = r + 8*half
                out[((size_t)bi * CC + c) * HW + m] = ac[r];
            }
        }
    }

    // ---- deterministic per-block Z partial ----
    zred[tid] = zloc;
    __syncthreads();
    for (int off = 64; off > 0; off >>= 1) {
        if (tid < off) zred[tid] += zred[tid + off];
        __syncthreads();
    }
    if (tid == 0) zpart[bi * gridDim.x + blockIdx.x] = zred[0];
}

// ---------------- Kernel 3: Z[b] = sum of 32 partials (deterministic) ----------------
__global__ __launch_bounds__(32) void pam_zreduce(const float* __restrict__ zpart,
                                                  float* __restrict__ Z) {
    __shared__ float red[32];
    red[threadIdx.x] = zpart[blockIdx.x * 32 + threadIdx.x];
    __syncthreads();
    for (int off = 16; off > 0; off >>= 1) {
        if (threadIdx.x < off) red[threadIdx.x] += red[threadIdx.x + off];
        __syncthreads();
    }
    if (threadIdx.x == 0) Z[blockIdx.x] = red[0];
}

// ---------------- Kernel 4: out /= Z[b] ----------------
__global__ __launch_bounds__(256) void pam_scale(float* __restrict__ out,
                                                 const float* __restrict__ Z) {
    int i = blockIdx.x * 256 + threadIdx.x;
    int b = i >> 18;  // 64*4096 = 2^18 elems per batch
    out[i] = out[i] / Z[b];
}

extern "C" void kernel_launch(void* const* d_in, const int* in_sizes, int n_in,
                              void* d_out, int out_size, void* d_ws, size_t ws_size,
                              hipStream_t stream) {
    const float* img = (const float*)d_in[0];
    float* out = (float*)d_out;

    float* Mw    = (float*)d_ws;   // [4]   per-batch max of diagonal (== max of S)
    float* Zw    = Mw + 4;         // [4]   per-batch Z
    float* zpart = Mw + 8;         // [128] per-block Z partials (32 blocks x 4 batches)

    hipMemsetAsync(Mw, 0, 4 * sizeof(float), stream);  // atomicMax identity (vals >= 0)

    pam_rowmax<<<dim3(HW / 256, 4), 256, 0, stream>>>(img, Mw);
    pam_main  <<<dim3(HW / MB, 4), 128, 0, stream>>>(img, out, Mw, zpart);
    pam_zreduce<<<4, 32, 0, stream>>>(zpart, Zw);
    pam_scale <<<(4 * CC * HW) / 256, 256, 0, stream>>>(out, Zw);
}
